// Linear8bit_66340064854179
// MI455X (gfx1250) — matrix operand order repaired and checked by reading, not compiled, not run on hardware
//
#include <hip/hip_runtime.h>

typedef __attribute__((ext_vector_type(16))) _Float16 v16h;
typedef __attribute__((ext_vector_type(8)))  float    v8f;
typedef __attribute__((ext_vector_type(4)))  float    f4;
typedef __attribute__((ext_vector_type(4)))  int      i4;
typedef __attribute__((ext_vector_type(4)))  unsigned int u4;   // 16 bytes
typedef __attribute__((ext_vector_type(4)))  _Float16 h4;       // 8 bytes

#define KDIM 4096
#define NDIM 16384
#define MDIM 4096
#define BM   128   // block tile M
#define BN   256   // block tile N
#define KB   64    // K-depth per LDS stage (2 WMMA k-steps)
#define LDSTR 72   // halfs per LDS row: 64 data + 8 pad (144B stride -> conflict-free b128 reads)

union frag32 { u4 q[2]; v16h h; };

// ---- x (f32) -> f16 ----------------------------------------------------
__global__ __launch_bounds__(256) void cvt_x_kernel(const float* __restrict__ x,
                                                    _Float16* __restrict__ out,
                                                    long long n) {
    long long i = ((long long)blockIdx.x * 256 + threadIdx.x) * 4;
    if (i + 3 < n) {
        f4 v = *(const f4*)(x + i);
        h4 o;
        o.x = (_Float16)v.x; o.y = (_Float16)v.y;
        o.z = (_Float16)v.z; o.w = (_Float16)v.w;
        *(h4*)(out + i) = o;
    }
}

// ---- W (u8 stored as i32) -> (q - zp) * scale, f16 ---------------------
__global__ __launch_bounds__(256) void dequant_w_kernel(const int* __restrict__ q,
                                                        const float* __restrict__ scale,
                                                        const float* __restrict__ zp,
                                                        _Float16* __restrict__ out,
                                                        long long n) {
    long long i = ((long long)blockIdx.x * 256 + threadIdx.x) * 4;
    if (i + 3 < n) {
        i4 v = *(const i4*)(q + i);
        float s = *scale, z = *zp;
        h4 o;
        o.x = (_Float16)(((float)v.x - z) * s);
        o.y = (_Float16)(((float)v.y - z) * s);
        o.z = (_Float16)(((float)v.z - z) * s);
        o.w = (_Float16)(((float)v.w - z) * s);
        *(h4*)(out + i) = o;
    }
}

// ---- GEMM: C[M,N] = A[M,K] * B[N,K]^T + bias ---------------------------
// Block: 128x256 tile, 256 threads (8 waves). Wave (wm, wn): 64x64 sub-tile
// = 4x4 f32 accumulators of 16x16 -> 4x fragment reuse, 1.0 ds_load_b128
// per WMMA. Ping-pong LDS: one barrier per 64-deep K stage; LDS stores for
// stage k+1 and global loads for stage k+2 overlap the 32 WMMAs of stage k.
__global__ __launch_bounds__(256) void gemm_f16_kernel(const _Float16* __restrict__ A,
                                                       const _Float16* __restrict__ B,
                                                       const float* __restrict__ bias,
                                                       float* __restrict__ C) {
    __shared__ _Float16 lA[2][BM * LDSTR];   // 2 x 18 KB
    __shared__ _Float16 lB[2][BN * LDSTR];   // 2 x 36 KB

    const int tid  = threadIdx.x;
    const int lane = tid & 31;
    const int wave = tid >> 5;        // 0..7
    const int wm   = wave & 1;        // 2 waves along M (64 rows each)
    const int wn   = wave >> 1;       // 4 waves along N (64 cols each)
    const int sub  = lane >> 4;       // half-wave selector
    const int l15  = lane & 15;

    const int m_base = blockIdx.y * BM;
    const int n_base = blockIdx.x * BN;

    // Global-load staging per 64-deep stage:
    //   A: thread t covers row (t>>1), halfs (t&1)*32 .. +31   (4 x 16B)
    //   B: thread t covers row t, halfs 0..63                  (8 x 16B)
    const int garow = tid >> 1;             // 0..127
    const int gacol = (tid & 1) * 32;       // half offset within 128B row

    const _Float16* __restrict__ pA = A + (long long)(m_base + garow) * KDIM + gacol;
    const _Float16* __restrict__ pB = B + (long long)(n_base + tid) * KDIM;
    _Float16* const sAw = &lA[0][garow * LDSTR + gacol];
    _Float16* const sBw = &lB[0][tid * LDSTR];
    const int bufA = BM * LDSTR;            // halfs between ping/pong A buffers
    const int bufB = BN * LDSTR;            // halfs between ping/pong B buffers

    u4 ra[4], rb[8];
    auto gload = [&](int kc) {
        const _Float16* qA = pA + (long long)kc * KB;
        const _Float16* qB = pB + (long long)kc * KB;
        #pragma unroll
        for (int i = 0; i < 4; ++i) ra[i] = *(const u4*)(qA + i * 8);
        #pragma unroll
        for (int i = 0; i < 8; ++i) rb[i] = *(const u4*)(qB + i * 8);
    };
    auto sstore = [&](int buf) {
        #pragma unroll
        for (int i = 0; i < 4; ++i) *(u4*)(sAw + buf * bufA + i * 8) = ra[i];
        #pragma unroll
        for (int i = 0; i < 8; ++i) *(u4*)(sBw + buf * bufB + i * 8) = rb[i];
    };

    v8f acc[4][4];
    #pragma unroll
    for (int i = 0; i < 4; ++i)
        #pragma unroll
        for (int j = 0; j < 4; ++j)
            acc[i][j] = (v8f)(0.f);

    const int nk = KDIM / KB;               // 64 stages

    // Prologue: stage 0 into buf 0, prefetch stage 1 into regs.
    gload(0);
    sstore(0);
    gload(1);
    __syncthreads();

    int cur = 0;
    for (int kc = 0; kc < nk; ++kc) {
        // Regs hold stage kc+1: store it into the idle buffer.
        if (kc + 1 < nk) sstore(cur ^ 1);
        // Prefetch stage kc+2 from global (overlaps WMMAs below).
        if (kc + 2 < nk) gload(kc + 2);

        const _Float16* sA = &lA[cur][0];
        const _Float16* sB = &lB[cur][0];

        #pragma unroll
        for (int ks = 0; ks < 2; ++ks) {
            // A fragments (16x32): lane l15 = row M; elements 0..7 = K[sub*8..],
            // elements 8..15 = K[16+sub*8..]   (ISA 7.12.2 16-bit A layout)
            v16h af[4];
            #pragma unroll
            for (int i = 0; i < 4; ++i) {
                const _Float16* p = sA + (wm * 64 + i * 16 + l15) * LDSTR + ks * 32 + sub * 8;
                frag32 f;
                f.q[0] = *(const u4*)(p);
                f.q[1] = *(const u4*)(p + 16);
                af[i] = f.h;
            }
            // B fragments (32x16): lane l15 = column N (= W row); elements 0..7 =
            // K[sub*8..], 8..15 = K[16+sub*8..]. One live at a time to bound register pressure.
            #pragma unroll
            for (int j = 0; j < 4; ++j) {
                const _Float16* p = sB + (wn * 64 + j * 16 + l15) * LDSTR + ks * 32 + sub * 8;
                frag32 f;
                f.q[0] = *(const u4*)(p);
                f.q[1] = *(const u4*)(p + 16);
                v16h bf = f.h;
                #pragma unroll
                for (int i = 0; i < 4; ++i)
                    acc[i][j] = __builtin_amdgcn_wmma_f32_16x16x32_f16(
                        /*neg_a=*/false, af[i], /*neg_b=*/false, bf,
                        /*c_mod=*/(short)0, acc[i][j],
                        /*reuse_a=*/false, /*reuse_b=*/false);
            }
        }

        __syncthreads();   // publish buf[cur^1]; all reads of buf[cur] done
        cur ^= 1;
    }

    // Epilogue: C/D layout (ISA 7.12.2): lane l15 = N, VGPR v = M row v + 8*sub
    #pragma unroll
    for (int j = 0; j < 4; ++j) {
        const int n = n_base + wn * 64 + j * 16 + l15;
        const float bv = bias[n];
        #pragma unroll
        for (int i = 0; i < 4; ++i) {
            #pragma unroll
            for (int v = 0; v < 8; ++v) {
                const int m = m_base + wm * 64 + i * 16 + v + 8 * sub;
                C[(long long)m * NDIM + n] = acc[i][j][v] + bv;
            }
        }
    }
}

extern "C" void kernel_launch(void* const* d_in, const int* in_sizes, int n_in,
                              void* d_out, int out_size, void* d_ws, size_t ws_size,
                              hipStream_t stream) {
    const float* x     = (const float*)d_in[0];
    const int*   w     = (const int*)  d_in[1];
    const float* scale = (const float*)d_in[2];
    const float* zp    = (const float*)d_in[3];
    const float* bias  = (const float*)d_in[4];
    float*       out   = (float*)d_out;

    _Float16* xh = (_Float16*)d_ws;                       // M*K halfs  (32 MB)
    _Float16* wh = xh + (size_t)MDIM * KDIM;              // N*K halfs  (128 MB)

    const long long nx = (long long)MDIM * KDIM;
    const long long nw = (long long)NDIM * KDIM;

    cvt_x_kernel<<<(unsigned)((nx / 4 + 255) / 256), 256, 0, stream>>>(x, xh, nx);
    dequant_w_kernel<<<(unsigned)((nw / 4 + 255) / 256), 256, 0, stream>>>(w, scale, zp, wh, nw);

    dim3 grid(NDIM / BN, MDIM / BM);   // 64 x 32 blocks
    gemm_f16_kernel<<<grid, 256, 0, stream>>>(xh, wh, bias, out);
}
